// Reversible_25735444038417
// MI455X (gfx1250) — matrix operand order repaired and checked by reading, not compiled, not run on hardware
//
#include <hip/hip_runtime.h>
#include <hip/hip_bf16.h>

// ---------------------------------------------------------------------------
// Reversible ODE solver, B=4096, D=512, HID=2048, 64 steps.
//   y1 = L*y0 + (1-L)*z0 + H*vf(t0, z0)
//   z1 = z0 + H*vf(t1, y1)
//   vf(t,y) = tanh(y@W1 + b1 + t*wt) @ W2 + b2
// ~2.2 TFLOP total, ~40 MB working set (L2-resident) -> compute-bound on the
// matrix pipes. bf16 WMMA (v_wmma_f32_16x16x32_bf16) with fp32 accumulation;
// all elementwise integration math stays fp32.
//
// GEMM scheme: 256-thread block (8 waves as 2Mx4N), block tile 128x128,
// K-chunk 32 staged through LDS (padded stride to spread banks). Each wave
// computes 64x32 (8 accumulators): 8 WMMAs per chunk from 6 LDS fragment
// reads; global traffic 16KB per 1.05 MFLOP per block-chunk (64 FLOP/B).
// ---------------------------------------------------------------------------

typedef __attribute__((ext_vector_type(16))) __bf16 v16bf;
typedef __attribute__((ext_vector_type(8)))  __bf16 bf16x8;
typedef __attribute__((ext_vector_type(8)))  float  v8f;

#define BATCH 4096
#define DIM   512
#define HID   2048
#define LCONST 0.999f
#define HCONST (1.0f / 64.0f)

// LDS tile: 128 rows x 32 K bf16, padded to 40 bf16 (80B) row stride so the
// 16B fragment reads from 16 consecutive rows hit disjoint bank groups.
#define LDS_STRIDE 40
#define LDS_TILE   (128 * LDS_STRIDE)

// Shared GEMM core. A: [*, K] bf16 row-major. Bt: [N, K] bf16 row-major
// (i.e. the transposed weight; each row of Bt is one B-matrix column).
// Computes the wave's 64x32 sub-tile of the 128x128 block tile into acc[8]
// (acc[i*2+j] = M-tile i (0..3), N-tile j (0..1)).
__device__ __forceinline__ void gemm_tiles(const __bf16* __restrict__ A,
                                           const __bf16* __restrict__ Bt,
                                           int K, int mBase, int nBase,
                                           __bf16* smem, v8f acc[8]) {
  const int tid  = threadIdx.x;
  const int lane = tid & 31;
  const int wave = tid >> 5;
  const int half = lane >> 4, l16 = lane & 15;
  const int wm = wave >> 2;   // 0..1 : M half of block tile
  const int wn = wave & 3;    // 0..3 : N quarter of block tile
  __bf16* As = smem;             // [128][LDS_STRIDE]
  __bf16* Bs = smem + LDS_TILE;  // [128][LDS_STRIDE]

  // Cooperative fill mapping: 512 16B segments per tile, 2 per thread.
  const int row0 = tid >> 1;                 // seg s = tid       -> row, seg
  const int seg0 = (tid & 1) * 2;            // covers segs {0,1} or {2,3}? no:
  // Simpler: segments s = tid and s + 256; row = s >> 2, seg = s & 3.

  for (int k0 = 0; k0 < K; k0 += 32) {
    __syncthreads();  // previous chunk's fragment reads done before overwrite
#pragma unroll
    for (int u = 0; u < 2; ++u) {
      const int s   = tid + u * 256;   // 0..511
      const int row = s >> 2;          // 0..127
      const int seg = s & 3;           // 16B segment within the 64B row
      const __bf16* ga = A  + (size_t)(mBase + row) * K + k0 + seg * 8;
      const __bf16* gb = Bt + (size_t)(nBase + row) * K + k0 + seg * 8;
      *(bf16x8*)&As[row * LDS_STRIDE + seg * 8] = *(const bf16x8*)ga;
      *(bf16x8*)&Bs[row * LDS_STRIDE + seg * 8] = *(const bf16x8*)gb;
      __builtin_prefetch(ga + 32, 0, 0);  // next K-chunk
      __builtin_prefetch(gb + 32, 0, 0);
    }
    __syncthreads();

    // B fragments for this wave's two N-tiles (ISA B layout: lane n holds
    // column N=n, K = half*8 + {0..7} in v0..3 and 16 + half*8 + {0..7} in v4..7).
    v16bf bfr[2];
#pragma unroll
    for (int j = 0; j < 2; ++j) {
      const __bf16* bp =
          &Bs[(wn * 32 + j * 16 + l16) * LDS_STRIDE + half * 8];
      ((bf16x8*)&bfr[j])[0] = *(const bf16x8*)bp;
      ((bf16x8*)&bfr[j])[1] = *(const bf16x8*)(bp + 16);
    }
    // A fragments (ISA A layout: lane m holds row M=m, K = half*8 + {0..7}
    // in v0..3 and 16 + half*8 + {0..7} in v4..7).
#pragma unroll
    for (int i = 0; i < 4; ++i) {
      const __bf16* ap =
          &As[(wm * 64 + i * 16 + l16) * LDS_STRIDE + half * 8];
      v16bf afr;
      ((bf16x8*)&afr)[0] = *(const bf16x8*)ap;
      ((bf16x8*)&afr)[1] = *(const bf16x8*)(ap + 16);
#pragma unroll
      for (int j = 0; j < 2; ++j) {
        acc[i * 2 + j] = __builtin_amdgcn_wmma_f32_16x16x32_bf16(
            false, afr, false, bfr[j], (short)0, acc[i * 2 + j], false, false);
      }
    }
  }
  (void)row0; (void)seg0;
}

// --- GEMM1 + bias + tanh:  Hid = tanh(A @ W1 + b1 + t*wt), bf16 out ---------
__global__ __launch_bounds__(256) void k_gemm1_tanh(
    const __bf16* __restrict__ Abf, const __bf16* __restrict__ W1T,
    const float* __restrict__ b1, const float* __restrict__ wt, float t,
    __bf16* __restrict__ Hid) {
  __shared__ __bf16 smem[2 * LDS_TILE];
  const int lane = threadIdx.x & 31;
  const int wave = threadIdx.x >> 5;
  const int half = lane >> 4, l16 = lane & 15;
  const int wm = wave >> 2, wn = wave & 3;
  const int mBase = blockIdx.y * 128;
  const int nBase = blockIdx.x * 128;

  v8f acc[8] = {};
  gemm_tiles(Abf, W1T, DIM, mBase, nBase, smem, acc);

#pragma unroll
  for (int j = 0; j < 2; ++j) {
    const int col = nBase + wn * 32 + j * 16 + l16;
    const float bias = b1[col] + t * wt[col];
#pragma unroll
    for (int i = 0; i < 4; ++i) {
#pragma unroll
      for (int r = 0; r < 8; ++r) {
        const int row = mBase + wm * 64 + i * 16 + half * 8 + r;
        Hid[(size_t)row * HID + col] = (__bf16)tanhf(acc[i * 2 + j][r] + bias);
      }
    }
  }
}

// --- GEMM2 + y-update: y = L*y + (1-L)*z + H*(Hid @ W2 + b2) ----------------
__global__ __launch_bounds__(256) void k_gemm2_y(
    const __bf16* __restrict__ Hid, const __bf16* __restrict__ W2T,
    const float* __restrict__ b2, const float* __restrict__ z,
    float* __restrict__ y, __bf16* __restrict__ Ybf) {
  __shared__ __bf16 smem[2 * LDS_TILE];
  const int lane = threadIdx.x & 31;
  const int wave = threadIdx.x >> 5;
  const int half = lane >> 4, l16 = lane & 15;
  const int wm = wave >> 2, wn = wave & 3;
  const int mBase = blockIdx.y * 128;
  const int nBase = blockIdx.x * 128;

  v8f acc[8] = {};
  gemm_tiles(Hid, W2T, HID, mBase, nBase, smem, acc);

#pragma unroll
  for (int j = 0; j < 2; ++j) {
    const int col = nBase + wn * 32 + j * 16 + l16;
    const float bias = b2[col];
#pragma unroll
    for (int i = 0; i < 4; ++i) {
#pragma unroll
      for (int r = 0; r < 8; ++r) {
        const int row = mBase + wm * 64 + i * 16 + half * 8 + r;
        const size_t idx = (size_t)row * DIM + col;
        const float yn = LCONST * y[idx] + (1.0f - LCONST) * z[idx] +
                         HCONST * (acc[i * 2 + j][r] + bias);
        y[idx] = yn;
        Ybf[idx] = (__bf16)yn;
      }
    }
  }
}

// --- GEMM2 + z-update: z = z + H*(Hid @ W2 + b2) ----------------------------
__global__ __launch_bounds__(256) void k_gemm2_z(
    const __bf16* __restrict__ Hid, const __bf16* __restrict__ W2T,
    const float* __restrict__ b2, float* __restrict__ z,
    __bf16* __restrict__ Zbf) {
  __shared__ __bf16 smem[2 * LDS_TILE];
  const int lane = threadIdx.x & 31;
  const int wave = threadIdx.x >> 5;
  const int half = lane >> 4, l16 = lane & 15;
  const int wm = wave >> 2, wn = wave & 3;
  const int mBase = blockIdx.y * 128;
  const int nBase = blockIdx.x * 128;

  v8f acc[8] = {};
  gemm_tiles(Hid, W2T, HID, mBase, nBase, smem, acc);

#pragma unroll
  for (int j = 0; j < 2; ++j) {
    const int col = nBase + wn * 32 + j * 16 + l16;
    const float bias = b2[col];
#pragma unroll
    for (int i = 0; i < 4; ++i) {
#pragma unroll
      for (int r = 0; r < 8; ++r) {
        const int row = mBase + wm * 64 + i * 16 + half * 8 + r;
        const size_t idx = (size_t)row * DIM + col;
        const float zn = z[idx] + HCONST * (acc[i * 2 + j][r] + bias);
        z[idx] = zn;
        Zbf[idx] = (__bf16)zn;
      }
    }
  }
}

// --- Weight prep: fp32 [R,C] row-major -> bf16 [C,R] row-major (transpose) --
__global__ void k_cvt_transpose(const float* __restrict__ src,
                                __bf16* __restrict__ dst, int R, int C) {
  const int idx = blockIdx.x * blockDim.x + threadIdx.x;
  if (idx < R * C) {
    const int r = idx / C, c = idx % C;
    dst[(size_t)c * R + r] = (__bf16)src[idx];
  }
}

// --- State init: y = z = y0 (fp32) and bf16 copies --------------------------
__global__ void k_init(const float* __restrict__ y0, float* __restrict__ y,
                       float* __restrict__ z, __bf16* __restrict__ ybf,
                       __bf16* __restrict__ zbf, int n) {
  const int idx = blockIdx.x * blockDim.x + threadIdx.x;
  if (idx < n) {
    const float v = y0[idx];
    y[idx] = v;
    z[idx] = v;
    const __bf16 b = (__bf16)v;
    ybf[idx] = b;
    zbf[idx] = b;
  }
}

extern "C" void kernel_launch(void* const* d_in, const int* in_sizes, int n_in,
                              void* d_out, int out_size, void* d_ws,
                              size_t ws_size, hipStream_t stream) {
  const float* y0 = (const float*)d_in[0];  // [4096, 512]
  const float* W1 = (const float*)d_in[1];  // [512, 2048]
  const float* b1 = (const float*)d_in[2];  // [2048]
  const float* wt = (const float*)d_in[3];  // [2048]
  const float* W2 = (const float*)d_in[4];  // [2048, 512]
  const float* b2 = (const float*)d_in[5];  // [512]
  float* y = (float*)d_out;                 // y state lives in d_out

  // Workspace layout (36 MB total):
  char* ws = (char*)d_ws;
  __bf16* W1T = (__bf16*)(ws);                        //  2 MB  [2048, 512]
  __bf16* W2T = (__bf16*)(ws + (2ull << 20));         //  2 MB  [512, 2048]
  __bf16* Hid = (__bf16*)(ws + (4ull << 20));         // 16 MB  [4096, 2048]
  __bf16* Ybf = (__bf16*)(ws + (20ull << 20));        //  4 MB  [4096, 512]
  __bf16* Zbf = (__bf16*)(ws + (24ull << 20));        //  4 MB  [4096, 512]
  float*  Z   = (float*)(ws + (28ull << 20));         //  8 MB  [4096, 512]

  const int nState = BATCH * DIM;
  const int nW = DIM * HID;

  k_cvt_transpose<<<(nW + 255) / 256, 256, 0, stream>>>(W1, W1T, DIM, HID);
  k_cvt_transpose<<<(nW + 255) / 256, 256, 0, stream>>>(W2, W2T, HID, DIM);
  k_init<<<(nState + 255) / 256, 256, 0, stream>>>(y0, y, Z, Ybf, Zbf, nState);

  const dim3 blk(256);
  const dim3 g1(HID / 128, BATCH / 128);  // 16 x 32 blocks, N=2048
  const dim3 g2(DIM / 128, BATCH / 128);  //  4 x 32 blocks, N=512

  for (int s = 0; s < 64; ++s) {
    const float t0 = (float)s * HCONST;
    const float t1 = (float)(s + 1) * HCONST;
    // y1 = L*y0 + (1-L)*z0 + H*vf(t0, z0)
    k_gemm1_tanh<<<g1, blk, 0, stream>>>(Zbf, W1T, b1, wt, t0, Hid);
    k_gemm2_y<<<g2, blk, 0, stream>>>(Hid, W2T, b2, Z, y, Ybf);
    // z1 = z0 + H*vf(t1, y1)
    k_gemm1_tanh<<<g1, blk, 0, stream>>>(Ybf, W1T, b1, wt, t1, Hid);
    k_gemm2_z<<<g2, blk, 0, stream>>>(Hid, W2T, b2, Z, Zbf);
  }
}
